// DKT2_47785806135753
// MI455X (gfx1250) — matrix operand order repaired and checked by reading, not compiled, not run on hardware
//
#include <hip/hip_runtime.h>
#include <hip/hip_bf16.h>

#define B_   16
#define SL_  512
#define S_   511
#define E_   256
#define NH_  4
#define I_   512
#define DHM_ 128
#define T_   (B_*S_)      // 8176 tokens, divisible by 16
#define F_   384
#define NSK_ 500

typedef __attribute__((ext_vector_type(16))) __bf16 v16bf;
typedef __attribute__((ext_vector_type(8)))  float  v8f;

// ---------------- WMMA fragment helpers (wave32, 16x16x32 bf16) ----------------
// A-frag: lane holds row M=lane&15; K elems k0 + (lane>>4)*16 + (0..15)
__device__ __forceinline__ v16bf fragA_f32(const float* __restrict__ row, int k0, int lane) {
  const int kh = ((lane >> 4) << 4);
  v16bf f;
#pragma unroll
  for (int j = 0; j < 8; ++j) f[j]     = (__bf16)row[k0 + kh + j];
#pragma unroll
  for (int j = 0; j < 8; ++j) f[8 + j] = (__bf16)row[k0 + 8 + kh + j];
  return f;
}
// B-frag: lane holds col N=lane&15; K elems k0 + (lane>>4)*16 + (0..15)
__device__ __forceinline__ v16bf fragB_bf16(const __bf16* __restrict__ row, int k0, int lane) {
  const int kb = ((lane >> 4) << 4);
  v16bf f;
#pragma unroll
  for (int j = 0; j < 16; ++j) f[j] = row[k0 + kb + j];
  return f;
}
__device__ __forceinline__ v16bf fragB_f32(const float* __restrict__ row, int k0, int lane) {
  const int kb = ((lane >> 4) << 4);
  v16bf f;
#pragma unroll
  for (int j = 0; j < 16; ++j) f[j] = (__bf16)row[k0 + kb + j];
  return f;
}

__device__ __forceinline__ float siluf(float v) { return v / (1.f + __expf(-v)); }
__device__ __forceinline__ float logsigf(float v) {
  return v >= 0.f ? -log1pf(__expf(-v)) : v - log1pf(__expf(v));
}

// ---------------- generic GEMM: C = act(A[M,K] * W[N,K]^T + bias) (+resid) ------
// block 256 = 8 waves (4 M-tiles x 2 N-tiles), block tile 64(M) x 32(N)
// Branch-free edge handling: out-of-range B rows are clamped; since a lane's B
// column feeds only that same lane's C column, the suppressed store makes it safe.
__global__ __launch_bounds__(256) void gemm_bf16_wmma(
    const float* __restrict__ A, long lda,
    const __bf16* __restrict__ W,
    const float* __restrict__ bias,
    const float* __restrict__ resid, long ldr,
    float* __restrict__ C, long ldc,
    int M, int N, int K, int act) {
  const int wave = threadIdx.x >> 5;
  const int lane = threadIdx.x & 31;
  const int m0 = blockIdx.y * 64 + (wave & 3) * 16;
  const int n0 = blockIdx.x * 32 + (wave >> 2) * 16;
  if (m0 >= M || n0 >= N) return;
  int arow = m0 + (lane & 15); if (arow >= M) arow = M - 1;
  const float* Arow = A + (long)arow * lda;
  const int nrow = n0 + (lane & 15);
  const __bf16* Wrow = W + (long)(nrow < N ? nrow : N - 1) * K;
  v8f acc = {0.f, 0.f, 0.f, 0.f, 0.f, 0.f, 0.f, 0.f};
  for (int k0 = 0; k0 < K; k0 += 32) {
    if (k0 + 64 < K) __builtin_prefetch(Arow + k0 + 64, 0, 1);  // global_prefetch
    v16bf a = fragA_f32(Arow, k0, lane);
    v16bf b = fragB_bf16(Wrow, k0, lane);
    acc = __builtin_amdgcn_wmma_f32_16x16x32_bf16(false, a, false, b, (short)0, acc,
                                                  false, false);
  }
  if (nrow >= N) return;
  const float bn = bias ? bias[nrow] : 0.f;
  const int mbase = m0 + ((lane >> 4) << 3);
#pragma unroll
  for (int r = 0; r < 8; ++r) {
    const int m = mbase + r;
    if (m >= M) continue;
    float v = acc[r] + bn;
    if (act == 1) v = fmaxf(v, 0.f);
    else if (act == 2) v = 1.f / (1.f + __expf(-v));
    if (resid) v += resid[(long)m * ldr + nrow];
    C[(long)m * ldc + nrow] = v;
  }
}

// ---------------- embeddings -----------------------------------------------------
__global__ void embed_kernel(const int* __restrict__ questions, const int* __restrict__ responses,
                             const int* __restrict__ skills,
                             const float* __restrict__ q_embed, const float* __restrict__ qa_embed,
                             const float* __restrict__ q_diff, const float* __restrict__ qa_diff,
                             const float* __restrict__ dparam,
                             float* __restrict__ x, float* __restrict__ qe,
                             float* __restrict__ pidb, int* __restrict__ tgtb,
                             int* __restrict__ qshb) {
  const long tok = blockIdx.x;
  const int b = (int)(tok / S_), s = (int)(tok % S_);
  const int pid = questions[b * SL_ + s];
  const int qd  = skills[b * SL_ + s];
  const int qs  = skills[b * SL_ + s + 1];
  const int r   = responses[b * SL_ + s];
  const int tg  = r > -1 ? r : 0;
  const float pe = dparam[pid];
  const int e = threadIdx.x;
  const float qev = q_embed[(long)qd * E_ + e];
  x[tok * E_ + e]  = qa_embed[(long)tg * E_ + e] + qev + pe * qa_diff[(long)tg * E_ + e];
  qe[tok * E_ + e] = qev + pe * q_diff[(long)qd * E_ + e];
  if (e == 0) { pidb[tok] = pe; tgtb[tok] = tg; qshb[tok] = qs; }
}

// ---------------- LayerNorm over rows of 256 (wave per row) ----------------------
__global__ void ln_rows(const float* __restrict__ x, const float* __restrict__ w,
                        float* __restrict__ out, long nrows) {
  const long row = (long)blockIdx.x * 8 + (threadIdx.x >> 5);
  if (row >= nrows) return;
  const int lane = threadIdx.x & 31;
  const float* xr = x + row * E_;
  float v[8]; float s = 0.f, s2 = 0.f;
#pragma unroll
  for (int j = 0; j < 8; ++j) { v[j] = xr[lane + j * 32]; s += v[j]; s2 += v[j] * v[j]; }
#pragma unroll
  for (int o = 16; o >= 1; o >>= 1) { s += __shfl_xor(s, o, 32); s2 += __shfl_xor(s2, o, 32); }
  const float mu = s * (1.f / E_);
  const float inv = rsqrtf(s2 * (1.f / E_) - mu * mu + 1e-5f);
#pragma unroll
  for (int j = 0; j < 8; ++j) {
    const int c = lane + j * 32;
    out[row * E_ + c] = (v[j] - mu) * inv * w[c];
  }
}

// ---------------- causal depthwise conv (K=4) + SiLU ----------------------------
__global__ void dwconv_silu(const float* __restrict__ x, long ldx,
                            const float* __restrict__ w, const float* __restrict__ bias,
                            float* __restrict__ out, long ldo, int C) {
  const long i = (long)blockIdx.x * 256 + threadIdx.x;
  if (i >= (long)T_ * C) return;
  const int c = (int)(i % C);
  const long tok = i / C;
  const int b = (int)(tok / S_), s = (int)(tok % S_);
  float acc = bias[c];
#pragma unroll
  for (int j = 0; j < 4; ++j) {
    const int sj = s + j - 3;
    if (sj >= 0) acc += w[c * 4 + j] * x[((long)b * S_ + sj) * ldx + c];
  }
  out[tok * ldo + c] = siluf(acc);
}

// ---------------- headwise 4x4 block mixes for q,k,v -----------------------------
__global__ void headwise_qkv(const float* __restrict__ xc, const float* __restrict__ xm, long ldxm,
                             const float* __restrict__ wq, const float* __restrict__ wk,
                             const float* __restrict__ wv,
                             float* __restrict__ q, float* __restrict__ k, float* __restrict__ v) {
  const long i = (long)blockIdx.x * 256 + threadIdx.x;
  if (i >= (long)T_ * 128) return;
  const int nb = (int)(i % 128);
  const long tok = i / 128;
  float xcv[4], xmv[4];
#pragma unroll
  for (int d = 0; d < 4; ++d) {
    xcv[d] = xc[tok * I_ + nb * 4 + d];
    xmv[d] = xm[tok * ldxm + nb * 4 + d];
  }
#pragma unroll
  for (int o = 0; o < 4; ++o) {
    float aq = 0.f, ak = 0.f, av = 0.f;
#pragma unroll
    for (int d = 0; d < 4; ++d) {
      const int wi = (nb * 4 + o) * 4 + d;
      aq += xcv[d] * wq[wi]; ak += xcv[d] * wk[wi]; av += xmv[d] * wv[wi];
    }
    q[tok * I_ + nb * 4 + o] = aq;
    k[tok * I_ + nb * 4 + o] = ak;
    v[tok * I_ + nb * 4 + o] = av;
  }
}

// ---------------- ig/fg head projections (dot over 1536) -------------------------
__global__ void igfg_kernel(const float* __restrict__ q, const float* __restrict__ k,
                            const float* __restrict__ v,
                            const float* __restrict__ wig, const float* __restrict__ big,
                            const float* __restrict__ wfg, const float* __restrict__ bfg,
                            float* __restrict__ ig, float* __restrict__ fg) {
  __shared__ float row[3 * I_];
  const long t = blockIdx.x;
  const int tid = threadIdx.x;
  for (int i = tid; i < I_; i += 256) {
    row[i]          = q[t * I_ + i];
    row[I_ + i]     = k[t * I_ + i];
    row[2 * I_ + i] = v[t * I_ + i];
  }
  __syncthreads();
  const int wave = tid >> 5, lane = tid & 31;
  const int h = wave & 3; const bool isf = wave >= 4;
  const float* w = (isf ? wfg : wig) + h * (3 * I_);
  float s = 0.f;
  for (int i = lane; i < 3 * I_; i += 32) s += row[i] * w[i];
#pragma unroll
  for (int o = 16; o >= 1; o >>= 1) s += __shfl_xor(s, o, 32);
  if (lane == 0) {
    const int b = (int)(t / S_), sl = (int)(t % S_);
    float* dst = isf ? fg : ig;
    dst[((long)b * NH_ + h) * S_ + sl] = s + (isf ? bfg[h] : big[h]);
  }
}

// ---------------- decay prefix scan: cs and maxD ---------------------------------
__global__ void decay_scan(const float* __restrict__ ig, const float* __restrict__ fg,
                           float* __restrict__ cs, float* __restrict__ mx) {
  const int idx = threadIdx.x;
  if (idx >= B_ * NH_) return;
  const float* f = fg + (long)idx * S_;
  const float* gi = ig + (long)idx * S_;
  float* c = cs + (long)idx * (S_ + 1);
  float* m = mx + (long)idx * S_;
  float acc = 0.f, P = -__builtin_inff();
  c[0] = 0.f;
  for (int s = 0; s < S_; ++s) {
    acc += logsigf(f[s]);
    c[s + 1] = acc;
    P = fmaxf(P, gi[s] - acc);
    m[s] = acc + P;  // maxD[t] = cs[t+1] + max_{s<=t}(ig[s]-cs[s+1])
  }
}

// ---------------- fused mLSTM attention (wave per 16-row tile per (b,h)) ---------
__global__ __launch_bounds__(32) void attn_mlstm(
    const float* __restrict__ q, const float* __restrict__ k, const float* __restrict__ v,
    const float* __restrict__ cs, const float* __restrict__ mx, const float* __restrict__ igv,
    float* __restrict__ hh) {
  const int tt = blockIdx.x, h = blockIdx.y, b = blockIdx.z;
  const int lane = threadIdx.x;
  const int lrow = lane & 15;
  const int kh = ((lane >> 4) << 3);
  const int mbase = ((lane >> 4) << 3);
  const int t0 = tt * 16;
  const long tokbase = (long)b * S_;
  const float* csbh = cs + ((long)b * NH_ + h) * (S_ + 1);
  const float* igbh = igv + ((long)b * NH_ + h) * S_;
  const float* mxbh = mx + ((long)b * NH_ + h) * S_;

  __shared__ __bf16 ctile[16][32];
  for (int i = lane; i < 256; i += 32) ctile[i >> 4][16 + (i & 15)] = (__bf16)0.f;
  __syncthreads();

  // q A-fragments for rows t0..t0+15, K = 128 in 4 steps
  int trow = t0 + lrow; if (trow >= S_) trow = S_ - 1;
  const float* qrow = q + (tokbase + trow) * (long)I_ + h * DHM_;
  v16bf fq[4];
#pragma unroll
  for (int kk = 0; kk < 4; ++kk) fq[kk] = fragA_f32(qrow, kk * 32, lane);

  v16bf zfrag;
#pragma unroll
  for (int j = 0; j < 16; ++j) zfrag[j] = (__bf16)0.f;

  float csrow[8], maxD[8], rs[8];
#pragma unroll
  for (int r = 0; r < 8; ++r) {
    int t = t0 + mbase + r; if (t >= S_) t = S_ - 1;
    csrow[r] = csbh[t + 1]; maxD[r] = mxbh[t]; rs[r] = 0.f;
  }
  v8f numer[8];
#pragma unroll
  for (int c = 0; c < 8; ++c) numer[c] = (v8f){0.f, 0.f, 0.f, 0.f, 0.f, 0.f, 0.f, 0.f};

  const float scale = 0.08838834764831845f;  // 1/sqrt(128)
  for (int st = 0; st <= tt; ++st) {
    const int s0 = st * 16;
    // qk^T tile via WMMA
    int scl = s0 + lrow; if (scl >= S_) scl = S_ - 1;
    const float* krow = k + (tokbase + scl) * (long)I_ + h * DHM_;
    v8f qk = {0.f, 0.f, 0.f, 0.f, 0.f, 0.f, 0.f, 0.f};
#pragma unroll
    for (int kk = 0; kk < 4; ++kk) {
      v16bf fk = fragB_f32(krow, kk * 32, lane);
      qk = __builtin_amdgcn_wmma_f32_16x16x32_bf16(false, fq[kk], false, fk, (short)0, qk,
                                                   false, false);
    }
    // decay * exp, rowsum partials, stash C tile to LDS (bf16)
    const int scol = s0 + (lane & 15);
    const float cscol = scol < S_ ? csbh[scol + 1] : 0.f;
    const float igcol = scol < S_ ? igbh[scol] : 0.f;
#pragma unroll
    for (int r = 0; r < 8; ++r) {
      const int t = t0 + mbase + r;
      float Cv = 0.f;
      if (scol <= t && scol < S_ && t < S_) {
        const float D = __expf(csrow[r] - cscol + igcol - maxD[r]);
        Cv = qk[r] * scale * D;
      }
      rs[r] += Cv;
      ctile[mbase + r][lane & 15] = (__bf16)Cv;
    }
    __syncthreads();
    // C A-fragment (16x32 with zero-padded upper K)
    v16bf fc;
#pragma unroll
    for (int j = 0; j < 8; ++j) { fc[j] = ctile[lrow][2 * kh + j]; fc[8 + j] = ctile[lrow][2 * kh + 8 + j]; }
    __syncthreads();
    // C @ V : 8 chunks of 16 columns of d
#pragma unroll
    for (int c = 0; c < 8; ++c) {
      const int dcol = c * 16 + (lane & 15);
      v16bf fv = zfrag;
      if (lane < 16) {
#pragma unroll
        for (int j = 0; j < 16; ++j) {
          const int s = s0 + j;
          fv[j] = (__bf16)(s < S_ ? v[(tokbase + s) * (long)I_ + h * DHM_ + dcol] : 0.f);
        }
      }
      numer[c] = __builtin_amdgcn_wmma_f32_16x16x32_bf16(false, fc, false, fv, (short)0,
                                                         numer[c], false, false);
    }
  }
  // reduce row-sums across the 16-lane group holding the same rows
#pragma unroll
  for (int o = 8; o >= 1; o >>= 1)
#pragma unroll
    for (int r = 0; r < 8; ++r) rs[r] += __shfl_xor(rs[r], o, 32);
  // normalize and store
#pragma unroll
  for (int r = 0; r < 8; ++r) {
    const int t = t0 + mbase + r;
    if (t >= S_) continue;
    const float nrm = fmaxf(fabsf(rs[r]), __expf(-maxD[r]));
    const float inv = 1.f / (nrm + 1e-6f);
    float* orow = hh + (tokbase + t) * (long)I_ + h * DHM_;
#pragma unroll
    for (int c = 0; c < 8; ++c) orow[c * 16 + (lane & 15)] = numer[c][r] * inv;
  }
}

// ---------------- group norms ----------------------------------------------------
__global__ void gn128_scale(float* __restrict__ d, const float* __restrict__ w) {
  const long grp = (long)blockIdx.x * 8 + (threadIdx.x >> 5);
  if (grp >= (long)T_ * NH_) return;
  const int lane = threadIdx.x & 31;
  float* p = d + grp * DHM_;
  float v[4]; float s = 0.f, s2 = 0.f;
#pragma unroll
  for (int j = 0; j < 4; ++j) { v[j] = p[lane + j * 32]; s += v[j]; s2 += v[j] * v[j]; }
#pragma unroll
  for (int o = 16; o >= 1; o >>= 1) { s += __shfl_xor(s, o, 32); s2 += __shfl_xor(s2, o, 32); }
  const float mu = s * (1.f / DHM_);
  const float inv = rsqrtf(s2 * (1.f / DHM_) - mu * mu + 1e-5f);
  const int h = (int)(grp & (NH_ - 1));
#pragma unroll
  for (int j = 0; j < 4; ++j) { const int c = lane + j * 32; p[c] = (v[j] - mu) * inv * w[h * DHM_ + c]; }
}

__global__ void gn64_res(const float* __restrict__ y, const float* __restrict__ w,
                         float* __restrict__ x) {
  const long grp = (long)blockIdx.x * 8 + (threadIdx.x >> 5);
  if (grp >= (long)T_ * NH_) return;
  const int lane = threadIdx.x & 31;
  const float* p = y + grp * 64;
  float v[2]; float s = 0.f, s2 = 0.f;
#pragma unroll
  for (int j = 0; j < 2; ++j) { v[j] = p[lane + j * 32]; s += v[j]; s2 += v[j] * v[j]; }
#pragma unroll
  for (int o = 16; o >= 1; o >>= 1) { s += __shfl_xor(s, o, 32); s2 += __shfl_xor(s2, o, 32); }
  const float mu = s * (1.f / 64.f);
  const float inv = rsqrtf(s2 * (1.f / 64.f) - mu * mu + 1e-5f);
  const long tok = grp >> 2; const int n = (int)(grp & 3);
#pragma unroll
  for (int j = 0; j < 2; ++j) {
    const int e = lane + j * 32;
    x[tok * E_ + n * 64 + e] += (v[j] - mu) * inv * w[n * 64 + e];
  }
}

// ---------------- misc pointwise -------------------------------------------------
__global__ void hout_kernel(const float* __restrict__ hh, const float* __restrict__ xc,
                            const float* __restrict__ up, const float* __restrict__ mskip,
                            float* __restrict__ out) {
  const long i = (long)blockIdx.x * 256 + threadIdx.x;
  if (i >= (long)T_ * I_) return;
  const int c = (int)(i % I_);
  const long tok = i / I_;
  const float z = up[tok * 1024 + I_ + c];
  out[i] = (hh[i] + mskip[c] * xc[i]) * siluf(z);
}

__global__ void ffnact_kernel(const float* __restrict__ f, float* __restrict__ out) {
  const long i = (long)blockIdx.x * 256 + threadIdx.x;
  if (i >= (long)T_ * F_) return;
  const int c = (int)(i % F_);
  const long tok = i / F_;
  out[i] = fmaxf(f[tok * (2 * F_) + c], 0.f) * f[tok * (2 * F_) + F_ + c];
}

__global__ void cq_kernel(const float* __restrict__ d, const float* __restrict__ qe,
                          const float* __restrict__ pidb, const int* __restrict__ tgtb,
                          float* __restrict__ cq) {
  const long i = (long)blockIdx.x * 256 + threadIdx.x;
  if (i >= (long)T_ * E_) return;
  const int c = (int)(i % E_);
  const long tok = i / E_;
  const float dv = d[tok * E_ + c];
  const int tg = tgtb[tok];
  float* row = cq + tok * 1024;
  row[c]           = dv - pidb[tok];
  row[E_ + c]      = qe[tok * E_ + c];
  row[2 * E_ + c]  = (tg == 1) ? dv : 0.f;
  row[3 * E_ + c]  = (tg == 0) ? dv : 0.f;
}

__global__ void gather_kernel(const float* __restrict__ probs, const int* __restrict__ qsh,
                              float* __restrict__ pred) {
  const long i = (long)blockIdx.x * 256 + threadIdx.x;
  if (i >= (long)T_) return;
  pred[i] = probs[i * 512 + qsh[i]];
}

// ---------------- weight prep ----------------------------------------------------
__global__ void f32_to_bf16(const float* __restrict__ in, __bf16* __restrict__ out, long n) {
  const long i = (long)blockIdx.x * 256 + threadIdx.x;
  if (i < n) out[i] = (__bf16)in[i];
}
// s_gate_w (g,n,d,e) -> bf16 (g,n,e,d); s_rec_w (g,n,d,e) -> bf16 (n,g,e,d)
__global__ void prep_gate_rec(const float* __restrict__ gate, const float* __restrict__ rec,
                              __bf16* __restrict__ gate_o, __bf16* __restrict__ rec_o) {
  const int i = blockIdx.x * 256 + threadIdx.x;
  if (i >= 4 * NH_ * 64 * 64) return;
  const int e = i & 63, d = (i >> 6) & 63, n = (i >> 12) & 3, g = (i >> 14) & 3;
  gate_o[(((g * NH_ + n) * 64 + e) << 6) + d] = (__bf16)gate[i];
  rec_o[(((n * 4 + g) * 64 + e) << 6) + d]    = (__bf16)rec[i];
}

// ---------------- sLSTM sequential scan (block per (b, head)) --------------------
__global__ __launch_bounds__(256) void slstm_scan(const float* __restrict__ wx,
                                                  const __bf16* __restrict__ wrec,
                                                  float* __restrict__ yout) {
  const int n = blockIdx.x, b = blockIdx.y;
  __shared__ __bf16 W[4 * 64 * 64];           // 32 KB of 320 KB WGP LDS
  __shared__ float ybuf[64], cbuf[64], nbuf[64], mbuf[64];
  __shared__ float raw[4 * 64];
  const int tid = threadIdx.x;
  const __bf16* wsrc = wrec + (long)n * 4 * 64 * 64;
  for (int i = tid; i < 4 * 64 * 64; i += 256) W[i] = wsrc[i];
  if (tid < 64) { ybuf[tid] = 0.f; cbuf[tid] = 0.f; nbuf[tid] = 0.f; mbuf[tid] = 0.f; }
  __syncthreads();
  const int g = tid >> 6, e = tid & 63;
  const __bf16* wge = W + (g * 64 + e) * 64;
  for (int t = 0; t < S_; ++t) {
    const long tok = (long)b * S_ + t;
    float acc = wx[tok * 1024 + g * E_ + n * 64 + e];
#pragma unroll 16
    for (int d = 0; d < 64; ++d) acc += ybuf[d] * (float)wge[d];
    raw[g * 64 + e] = acc;
    __syncthreads();
    if (tid < 64) {
      const float ir = raw[tid], fr = raw[64 + tid], zr = raw[128 + tid], orr = raw[192 + tid];
      const float lfm = mbuf[tid] + logsigf(fr);
      const float mn = fmaxf(ir, lfm);
      const float igx = __expf(ir - mn);
      const float fgx = __expf(lfm - mn);
      const float cn = fgx * cbuf[tid] + igx * tanhf(zr);
      const float nn = fgx * nbuf[tid] + igx;
      const float yv = (1.f / (1.f + __expf(-orr))) * cn / nn;
      cbuf[tid] = cn; nbuf[tid] = nn; mbuf[tid] = mn; ybuf[tid] = yv;
      yout[tok * E_ + n * 64 + tid] = yv;
    }
    __syncthreads();
  }
}

// =================================================================================
static inline int cdiv(long a, long b) { return (int)((a + b - 1) / b); }

extern "C" void kernel_launch(void* const* d_in, const int* in_sizes, int n_in,
                              void* d_out, int out_size, void* d_ws, size_t ws_size,
                              hipStream_t stream) {
  const int*   questions = (const int*)d_in[0];
  const int*   responses = (const int*)d_in[1];
  const int*   skills    = (const int*)d_in[2];
  const float* q_embed   = (const float*)d_in[4];
  const float* qa_embed  = (const float*)d_in[5];
  const float* q_diff    = (const float*)d_in[6];
  const float* qa_diff   = (const float*)d_in[7];
  const float* dparam    = (const float*)d_in[8];
  const float* ln0_w     = (const float*)d_in[9];
  const float* m_up_w    = (const float*)d_in[10];
  const float* m_up_b    = (const float*)d_in[11];
  const float* m_conv_w  = (const float*)d_in[12];
  const float* m_conv_b  = (const float*)d_in[13];
  const float* m_q_w     = (const float*)d_in[14];
  const float* m_k_w     = (const float*)d_in[15];
  const float* m_v_w     = (const float*)d_in[16];
  const float* m_ig_w    = (const float*)d_in[17];
  const float* m_ig_b    = (const float*)d_in[18];
  const float* m_fg_w    = (const float*)d_in[19];
  const float* m_fg_b    = (const float*)d_in[20];
  const float* m_outnorm = (const float*)d_in[21];
  const float* m_skip    = (const float*)d_in[22];
  const float* m_down_w  = (const float*)d_in[23];
  const float* m_down_b  = (const float*)d_in[24];
  const float* ln1_w     = (const float*)d_in[25];
  const float* s_conv_w  = (const float*)d_in[26];
  const float* s_conv_b  = (const float*)d_in[27];
  const float* s_gate_w  = (const float*)d_in[28];
  const float* s_rec_w   = (const float*)d_in[29];
  const float* s_bias    = (const float*)d_in[30];
  const float* s_gn_w    = (const float*)d_in[31];
  const float* ln2_w     = (const float*)d_in[32];
  const float* ffn_up_w  = (const float*)d_in[33];
  const float* ffn_up_b  = (const float*)d_in[34];
  const float* ffn_dn_w  = (const float*)d_in[35];
  const float* ffn_dn_b  = (const float*)d_in[36];
  const float* post_w    = (const float*)d_in[37];
  const float* out_w1    = (const float*)d_in[38];
  const float* out_b1    = (const float*)d_in[39];
  const float* out_w2    = (const float*)d_in[40];
  const float* out_b2    = (const float*)d_in[41];
  const float* out_w3    = (const float*)d_in[42];
  const float* out_b3    = (const float*)d_in[43];
  float* pred = (float*)d_out;

  // ---- workspace carve ----
  char* wp = (char*)d_ws;
  auto alloc = [&](size_t bytes) -> void* {
    void* p = (void*)wp; wp += (bytes + 255) & ~(size_t)255; return p;
  };
  int*   tgtb = (int*)alloc((size_t)T_ * 4);
  int*   qshb = (int*)alloc((size_t)T_ * 4);
  float* pidb = (float*)alloc((size_t)T_ * 4);
  float* qe   = (float*)alloc((size_t)T_ * E_ * 4);
  float* x    = (float*)alloc((size_t)T_ * E_ * 4);
  float* h    = (float*)alloc((size_t)T_ * E_ * 4);
  float* up   = (float*)alloc((size_t)T_ * 1024 * 4);
  float* xc   = (float*)alloc((size_t)T_ * I_ * 4);
  float* qb   = (float*)alloc((size_t)T_ * I_ * 4);
  float* kb   = (float*)alloc((size_t)T_ * I_ * 4);
  float* vb   = (float*)alloc((size_t)T_ * I_ * 4);
  float* hhb  = (float*)alloc((size_t)T_ * I_ * 4);
  float* wx   = (float*)alloc((size_t)T_ * 1024 * 4);
  float* hc   = (float*)alloc((size_t)T_ * E_ * 4);
  float* db   = (float*)alloc((size_t)T_ * E_ * 4);
  float* o2   = (float*)alloc((size_t)T_ * E_ * 4);
  float* igb  = (float*)alloc((size_t)B_ * NH_ * S_ * 4);
  float* fgb  = (float*)alloc((size_t)B_ * NH_ * S_ * 4);
  float* csb  = (float*)alloc((size_t)B_ * NH_ * (S_ + 1) * 4);
  float* mxb  = (float*)alloc((size_t)B_ * NH_ * S_ * 4);
  __bf16* up_bf  = (__bf16*)alloc((size_t)1024 * 256 * 2);
  __bf16* dn_bf  = (__bf16*)alloc((size_t)256 * 512 * 2);
  __bf16* fu_bf  = (__bf16*)alloc((size_t)768 * 256 * 2);
  __bf16* fd_bf  = (__bf16*)alloc((size_t)256 * 384 * 2);
  __bf16* w1_bf  = (__bf16*)alloc((size_t)512 * 1024 * 2);
  __bf16* w2_bf  = (__bf16*)alloc((size_t)256 * 512 * 2);
  __bf16* w3_bf  = (__bf16*)alloc((size_t)NSK_ * 256 * 2);
  __bf16* gate_bf = (__bf16*)alloc((size_t)4 * NH_ * 64 * 64 * 2);
  __bf16* rec_bf  = (__bf16*)alloc((size_t)4 * NH_ * 64 * 64 * 2);
  // dead-range aliases
  float* hout   = vb;   // v dead after attention + ig/fg
  float* cq     = up;   // up dead after hout
  float* ffnb   = wx;   // wx dead after scan
  float* ffnact = kb;   // k dead after attention
  float* probs  = qb;   // q dead after attention (ldc 512, N 500)
  float* yb     = hhb;  // hh dead after hout
  float* o1     = xc;   // xc dead after down-proj

  // ---- weight prep (bf16) ----
  auto cvt = [&](const float* src, __bf16* dst, long n) {
    f32_to_bf16<<<cdiv(n, 256), 256, 0, stream>>>(src, dst, n);
  };
  cvt(m_up_w, up_bf, 1024L * 256);
  cvt(m_down_w, dn_bf, 256L * 512);
  cvt(ffn_up_w, fu_bf, 768L * 256);
  cvt(ffn_dn_w, fd_bf, 256L * 384);
  cvt(out_w1, w1_bf, 512L * 1024);
  cvt(out_w2, w2_bf, 256L * 512);
  cvt(out_w3, w3_bf, (long)NSK_ * 256);
  prep_gate_rec<<<cdiv(4 * NH_ * 64 * 64, 256), 256, 0, stream>>>(s_gate_w, s_rec_w,
                                                                  gate_bf, rec_bf);

  auto gemm = [&](const float* A, long lda, const __bf16* W, const float* bias,
                  const float* resid, long ldr, float* C, long ldc, int N, int K, int act) {
    dim3 grid(cdiv(N, 32), cdiv(T_, 64));
    gemm_bf16_wmma<<<grid, 256, 0, stream>>>(A, lda, W, bias, resid, ldr, C, ldc,
                                             T_, N, K, act);
  };

  // ---- stage 0: embeddings + LN0 + up-proj ----
  embed_kernel<<<T_, 256, 0, stream>>>(questions, responses, skills, q_embed, qa_embed,
                                       q_diff, qa_diff, dparam, x, qe, pidb, tgtb, qshb);
  ln_rows<<<cdiv(T_, 8), 256, 0, stream>>>(x, ln0_w, h, T_);
  gemm(h, E_, up_bf, m_up_b, nullptr, 0, up, 1024, 1024, E_, 0);

  // ---- mLSTM branch ----
  dwconv_silu<<<cdiv((long)T_ * I_, 256), 256, 0, stream>>>(up, 1024, m_conv_w, m_conv_b,
                                                            xc, I_, I_);
  headwise_qkv<<<cdiv((long)T_ * 128, 256), 256, 0, stream>>>(xc, up, 1024, m_q_w, m_k_w,
                                                              m_v_w, qb, kb, vb);
  igfg_kernel<<<T_, 256, 0, stream>>>(qb, kb, vb, m_ig_w, m_ig_b, m_fg_w, m_fg_b, igb, fgb);
  decay_scan<<<1, 64, 0, stream>>>(igb, fgb, csb, mxb);
  attn_mlstm<<<dim3(32, NH_, B_), 32, 0, stream>>>(qb, kb, vb, csb, mxb, igb, hhb);
  gn128_scale<<<cdiv((long)T_ * NH_, 8), 256, 0, stream>>>(hhb, m_outnorm);
  hout_kernel<<<cdiv((long)T_ * I_, 256), 256, 0, stream>>>(hhb, xc, up, m_skip, hout);
  gemm(hout, I_, dn_bf, m_down_b, x, E_, x, E_, E_, I_, 0);

  // ---- sLSTM branch ----
  ln_rows<<<cdiv(T_, 8), 256, 0, stream>>>(x, ln1_w, h, T_);
  dwconv_silu<<<cdiv((long)T_ * E_, 256), 256, 0, stream>>>(h, E_, s_conv_w, s_conv_b,
                                                            hc, E_, E_);
  for (int g = 0; g < 4; ++g)
    for (int n = 0; n < NH_; ++n) {
      const float* Ain = (g < 2 ? hc : h) + n * 64;
      dim3 grid(2, cdiv(T_, 64));
      gemm_bf16_wmma<<<grid, 256, 0, stream>>>(Ain, E_, gate_bf + (size_t)(g * NH_ + n) * 4096,
                                               s_bias + (g * NH_ + n) * 64, nullptr, 0,
                                               wx + g * E_ + n * 64, 1024, T_, 64, 64, 0);
    }
  slstm_scan<<<dim3(NH_, B_), 256, 0, stream>>>(wx, rec_bf, yb);
  gn64_res<<<cdiv((long)T_ * NH_, 8), 256, 0, stream>>>(yb, s_gn_w, x);

  // ---- FFN ----
  ln_rows<<<cdiv(T_, 8), 256, 0, stream>>>(x, ln2_w, h, T_);
  gemm(h, E_, fu_bf, ffn_up_b, nullptr, 0, ffnb, 2 * F_, 2 * F_, E_, 0);
  ffnact_kernel<<<cdiv((long)T_ * F_, 256), 256, 0, stream>>>(ffnb, ffnact);
  gemm(ffnact, F_, fd_bf, ffn_dn_b, x, E_, x, E_, E_, F_, 0);

  // ---- output head ----
  ln_rows<<<cdiv(T_, 8), 256, 0, stream>>>(x, post_w, db, T_);
  cq_kernel<<<cdiv((long)T_ * E_, 256), 256, 0, stream>>>(db, qe, pidb, tgtb, cq);
  gemm(cq, 1024, w1_bf, out_b1, nullptr, 0, o1, 512, 512, 1024, 1);
  gemm(o1, 512, w2_bf, out_b2, nullptr, 0, o2, E_, E_, 512, 1);
  gemm(o2, E_, w3_bf, out_b3, nullptr, 0, probs, 512, NSK_, E_, 2);
  gather_kernel<<<cdiv(T_, 256), 256, 0, stream>>>(probs, qshb, pred);
}
